// Codebook_68951404970007
// MI455X (gfx1250) — matrix operand order repaired and checked by reading, not compiled, not run on hardware
//
#include <hip/hip_runtime.h>

// ---------------------------------------------------------------------------
// Fused VQ codebook kernel for gfx1250 (MI455X):
//   dists = x @ cb^T  (bf16 WMMA, f32 accum) -> fused argmin -> gather ->
//   quantize write + loss.  Never materializes the [N,K] dists matrix.
// v2: 32 rows/wave (each LDS B-fragment feeds 2 WMMAs; two independent
//     accumulator chains hide WMMA latency) + double-buffered codebook
//     staging (global reads overlap the WMMA stream).
// ---------------------------------------------------------------------------

static constexpr int DIM    = 256;     // latent dim
static constexpr int NCODES = 1024;    // codebook size
static constexpr int NTOK   = 65536;   // tokens
static constexpr int TILE   = 128;     // codes per LDS tile (x2 buffers)
static constexpr int PITCH  = 264;     // bf16 elems per LDS row (528 B: 16B-aligned,
                                       // lane stride = 132 dwords = 4 banks -> no conflicts)

typedef __attribute__((ext_vector_type(16))) __bf16 v16bf;
typedef __attribute__((ext_vector_type(8)))  __bf16 v8bf;
typedef __attribute__((ext_vector_type(4)))  __bf16 v4bf;
typedef __attribute__((ext_vector_type(8)))  float  v8f;
typedef __attribute__((ext_vector_type(4)))  float  v4f;

__global__ void vq_zero_loss(float* loss) { *loss = 0.0f; }

__device__ __forceinline__ void stage_tile(__bf16* __restrict__ dst,
                                           const float* __restrict__ cb,
                                           int tile, int tid)
{
    const float* src = cb + (size_t)tile * TILE * DIM;
#pragma unroll 4
    for (int k = 0; k < (TILE * DIM) / (4 * 256); ++k) {   // 32 float4 / thread
        int f = k * 256 + tid;                              // float4 index in tile
        v4f v = *(const v4f*)(src + (size_t)f * 4);
        v4bf b;
        b[0] = (__bf16)v[0]; b[1] = (__bf16)v[1];
        b[2] = (__bf16)v[2]; b[3] = (__bf16)v[3];
        int r  = f >> 6;                                    // row (64 float4 per row)
        int c4 = f & 63;
        *(v4bf*)(dst + r * PITCH + c4 * 4) = b;
    }
}

__global__ __launch_bounds__(256)
void vq_fused(const float* __restrict__ x, const float* __restrict__ cb,
              float* __restrict__ outq, float* __restrict__ outloss,
              float* __restrict__ outidx)
{
    __shared__ __bf16 lcb[2 * TILE * PITCH];       // 2 x 67584 B = 135168 B

    const int tid  = threadIdx.x;
    const int wave = tid >> 5;
    const int lane = tid & 31;
    const int half = lane >> 4;                    // wave32: two 16-lane halves
    const int m    = lane & 15;
    const int row0 = (blockIdx.x * 8 + wave) * 32; // 32 rows per wave
    const int NT   = NCODES / TILE;                // 8 code tiles

    // ---- A fragments: two 16x256 x-strips, bf16, resident in VGPRs ----
    // 16-bit A 16x32 layout: lane(half0) holds K = {0..7} u {16..23} (+chunk*32),
    // lane(half1) holds K = {8..15} u {24..31}  -> two contiguous 8-float runs.
    v16bf a[2][8];
#pragma unroll
    for (int s = 0; s < 2; ++s) {
        const int row = row0 + s * 16 + m;
#pragma unroll
        for (int c = 0; c < 8; ++c) {
            const float* px = x + (size_t)row * DIM + c * 32 + half * 8;
            v4f f0 = *(const v4f*)(px + 0);
            v4f f1 = *(const v4f*)(px + 4);
            v4f f2 = *(const v4f*)(px + 16);
            v4f f3 = *(const v4f*)(px + 20);
#pragma unroll
            for (int e = 0; e < 4; ++e) {
                a[s][c][e]      = (__bf16)f0[e];
                a[s][c][e + 4]  = (__bf16)f1[e];
                a[s][c][e + 8]  = (__bf16)f2[e];
                a[s][c][e + 12] = (__bf16)f3[e];
            }
        }
    }

    // ---- running (min, argmin): lane owns column residue (lane&15);
    //      component j owns row j + 8*half of each strip ----
    float    minv[2][8];
    unsigned mini[2][8];
#pragma unroll
    for (int s = 0; s < 2; ++s)
#pragma unroll
        for (int j = 0; j < 8; ++j) { minv[s][j] = __builtin_inff(); mini[s][j] = 0u; }

    // ---- software pipeline: stage tile 0, then overlap stage(ct+1)/compute(ct)
    stage_tile(lcb, cb, 0, tid);
    __syncthreads();

    for (int ct = 0; ct < NT; ++ct) {
        if (ct + 1 < NT)
            stage_tile(lcb + ((ct + 1) & 1) * (TILE * PITCH), cb, ct + 1, tid);
        if (ct + 2 < NT)   // warm L2 one more tile ahead (global_prefetch_b8)
            __builtin_prefetch(cb + (size_t)(ct + 2) * TILE * DIM + tid * 32, 0, 1);

        const __bf16* buf = lcb + (ct & 1) * (TILE * PITCH);

#pragma unroll 1
        for (int t = 0; t < TILE / 16; ++t) {      // 8 column tiles of 16 codes
            v8f acc0 = {0.f,0.f,0.f,0.f,0.f,0.f,0.f,0.f};
            v8f acc1 = {0.f,0.f,0.f,0.f,0.f,0.f,0.f,0.f};
            // 16-bit B 32x16 layout: lane holds col = lane&15, K = {8*half..+7}
            // u {16+8*half..+7} within each 32-chunk (matches A).
            const __bf16* pb = buf + (t * 16 + m) * PITCH + half * 8;
#pragma unroll
            for (int c = 0; c < 8; ++c) {          // D = 8 chunks of 32
                v8bf b0 = *(const v8bf*)(pb + c * 32);
                v8bf b1 = *(const v8bf*)(pb + c * 32 + 16);
                v16bf bb = __builtin_shufflevector(b0, b1,
                    0,1,2,3,4,5,6,7,8,9,10,11,12,13,14,15);
                acc0 = __builtin_amdgcn_wmma_f32_16x16x32_bf16(
                    false, a[0][c], false, bb, (short)0, acc0, false, false);
                acc1 = __builtin_amdgcn_wmma_f32_16x16x32_bf16(
                    false, a[1][c], false, bb, (short)0, acc1, false, false);
            }
            unsigned kidx = (unsigned)(ct * TILE + t * 16 + m);
#pragma unroll
            for (int j = 0; j < 8; ++j) {
                if (acc0[j] < minv[0][j]) { minv[0][j] = acc0[j]; mini[0][j] = kidx; }
                if (acc1[j] < minv[1][j]) { minv[1][j] = acc1[j]; mini[1][j] = kidx; }
            }
        }
        __syncthreads();   // stage(ct+1) visible; compute(ct) done before restage
    }

    // ---- argmin reduction across the 16 lanes of each half (wave32 shfl) ----
#pragma unroll
    for (int off = 1; off <= 8; off <<= 1) {
#pragma unroll
        for (int s = 0; s < 2; ++s)
#pragma unroll
            for (int j = 0; j < 8; ++j) {
                float    ov = __shfl_xor(minv[s][j], off, 32);
                unsigned oi = __shfl_xor(mini[s][j], off, 32);
                if (ov < minv[s][j] || (ov == minv[s][j] && oi < mini[s][j])) {
                    minv[s][j] = ov; mini[s][j] = oi;
                }
            }
    }

    unsigned idx_all[32];                          // per-row winners, all lanes
#pragma unroll
    for (int s = 0; s < 2; ++s)
#pragma unroll
        for (int j = 0; j < 8; ++j) {
            idx_all[s * 16 + j]     = __shfl(mini[s][j], 0, 32);   // rows 0..7
            idx_all[s * 16 + j + 8] = __shfl(mini[s][j], 16, 32);  // rows 8..15
        }

    if (lane == 0) {
#pragma unroll
        for (int r = 0; r < 32; ++r)
            outidx[row0 + r] = (float)idx_all[r];
    }

    // ---- gather codebook rows (fp32, L2-hot), write quantize, loss ----
    float lsum = 0.0f;
#pragma unroll
    for (int r = 0; r < 32; ++r) {
        const float* crow = cb   + (size_t)idx_all[r] * DIM;
        const float* xrow = x    + (size_t)(row0 + r) * DIM;
        float*       qrow = outq + (size_t)(row0 + r) * DIM;
        const int d0 = lane * 8;                   // 32 lanes x 8 floats = 256
        v4f q0 = *(const v4f*)(crow + d0);
        v4f q1 = *(const v4f*)(crow + d0 + 4);
        v4f x0 = *(const v4f*)(xrow + d0);
        v4f x1 = *(const v4f*)(xrow + d0 + 4);
        *(v4f*)(qrow + d0)     = q0;
        *(v4f*)(qrow + d0 + 4) = q1;
#pragma unroll
        for (int e = 0; e < 4; ++e) {
            float e0 = q0[e] - x0[e];
            float e1 = q1[e] - x1[e];
            lsum += e0 * e0 + e1 * e1;
        }
    }
#pragma unroll
    for (int off = 1; off < 32; off <<= 1)
        lsum += __shfl_xor(lsum, off, 32);
    if (lane == 0)                                 // loss = (1+BETA)*mean(err^2)
        atomicAdd(outloss, lsum * (1.25f / 16777216.0f));
}

extern "C" void kernel_launch(void* const* d_in, const int* in_sizes, int n_in,
                              void* d_out, int out_size, void* d_ws, size_t ws_size,
                              hipStream_t stream)
{
    const float* x  = (const float*)d_in[0];   // [65536, 256]
    const float* cb = (const float*)d_in[1];   // [1024, 256]

    float* outq    = (float*)d_out;                    // quantize [N*D]
    float* outloss = outq + (size_t)NTOK * DIM;        // scalar loss
    float* outidx  = outloss + 1;                      // idx [N] (as float)

    vq_zero_loss<<<1, 1, 0, stream>>>(outloss);
    vq_fused<<<NTOK / 256, 256, 0, stream>>>(x, cb, outq, outloss, outidx);
}
